// AttentionHead_9483287789912
// MI455X (gfx1250) — matrix operand order repaired and checked by reading, not compiled, not run on hardware
//
#include <hip/hip_runtime.h>

typedef __attribute__((ext_vector_type(16))) _Float16 v16h;
typedef __attribute__((ext_vector_type(8)))  _Float16 v8h;
typedef __attribute__((ext_vector_type(8)))  float    v8f;
typedef __attribute__((ext_vector_type(4)))  float    v4f;

// Scheduling fence: no instruction may be reordered across this point.
// Used to keep load groups batched ahead of WMMA groups (the machine
// scheduler otherwise sinks each load pair to its consumer, serializing).
#define SCHED_FENCE() __builtin_amdgcn_sched_barrier(0)

static __device__ __forceinline__ v8f wmma_f16(v16h a, v16h b, v8f c) {
  // D = A(16x32 f16) * B(32x16 f16) + C(16x16 f32)
  return __builtin_amdgcn_wmma_f32_16x16x32_f16(false, a, false, b, (short)0, c,
                                                false, false);
}

static __device__ __forceinline__ v16h cat8(v8h lo8, v8h hi8) {
  return __builtin_shufflevector(lo8, hi8, 0, 1, 2, 3, 4, 5, 6, 7, 8, 9, 10, 11,
                                 12, 13, 14, 15);
}

// ---------------------------------------------------------------------------
// Kernel 0: bulk f32 -> f16 conversion (bandwidth-bound, one pass).
// ---------------------------------------------------------------------------
__global__ __launch_bounds__(256) void cvt_f16_kernel(
    const float* __restrict__ in, _Float16* __restrict__ out, int n8) {
  const int idx = blockIdx.x * 256 + threadIdx.x;
  const int stride = gridDim.x * 256;
  for (int i = idx; i < n8; i += stride) {
    v4f f0 = *(const v4f*)(in + (size_t)i * 8);
    v4f f1 = *(const v4f*)(in + (size_t)i * 8 + 4);
    v8h h;
#pragma unroll
    for (int j = 0; j < 4; ++j) {
      h[j] = (_Float16)f0[j];
      h[j + 4] = (_Float16)f1[j];
    }
    *(v8h*)(out + (size_t)i * 8) = h;
  }
}

// ---------------------------------------------------------------------------
// Kernel 1: QKV projection, all-f16 operands, 2x-unrolled ping-pong pipeline
// with sched_barrier-pinned load/WMMA groups. One wave -> one 16-row tile of
// one matrix, 4 col-tile accumulators (16x64). Q,K stored f16 row-major
// [B*N,64]; V stored transposed [B,64,N].
// ---------------------------------------------------------------------------
struct PFrag {
  v16h a;
  v16h b[4];
};

static __device__ __forceinline__ void load_pfrag(const _Float16* ap,
                                                  const _Float16* bp, int kk,
                                                  int kbase, PFrag& f) {
  f.a = cat8(*(const v8h*)(ap + kk + kbase),
             *(const v8h*)(ap + kk + 8 + kbase));
#pragma unroll
  for (int ct = 0; ct < 4; ++ct)
    f.b[ct] = *(const v16h*)(bp + (size_t)ct * 16 * 1024 + kk);
}

__global__ __launch_bounds__(256) void qkv_proj_kernel(
    const _Float16* __restrict__ xh, const _Float16* __restrict__ wq,
    const _Float16* __restrict__ wk, const _Float16* __restrict__ wv,
    _Float16* __restrict__ qh, _Float16* __restrict__ kh,
    _Float16* __restrict__ vt) {
  const int wave = threadIdx.x >> 5;
  const int lane = threadIdx.x & 31;
  const int hi = lane >> 4, lo = lane & 15;
  const int kbase = hi * 16;
  const int rt = blockIdx.x * 8 + wave;  // 0..1023 row tiles of 16
  const int mat = blockIdx.y;            // 0:Q 1:K 2:V
  const _Float16* W = (mat == 0) ? wq : (mat == 1) ? wk : wv;

  const _Float16* ap = xh + (size_t)(rt * 16 + lo) * 1024;
  const _Float16* bp = W + (size_t)lo * 1024 + hi * 16;

  v8f c[4] = {};

  PFrag fA, fB;
  load_pfrag(ap, bp, 0, kbase, fA);

  for (int kk = 0; kk < 1024; kk += 64) {
    load_pfrag(ap, bp, kk + 32, kbase, fB);  // buffer B: chunk kk+32
    SCHED_FENCE();
#pragma unroll
    for (int ct = 0; ct < 4; ++ct) c[ct] = wmma_f16(fA.a, fA.b[ct], c[ct]);
    const int kn2 = (kk + 64 < 1024) ? kk + 64 : 0;  // last load unused
    load_pfrag(ap, bp, kn2, kbase, fA);      // buffer A: chunk kk+64
    SCHED_FENCE();
#pragma unroll
    for (int ct = 0; ct < 4; ++ct) c[ct] = wmma_f16(fB.a, fB.b[ct], c[ct]);
  }

  // D layout: reg r -> row = r + 8*hi, col = lane&15
#pragma unroll
  for (int ct = 0; ct < 4; ++ct) {
#pragma unroll
    for (int r = 0; r < 8; ++r) {
      const int grow = rt * 16 + r + hi * 8;  // global row in [0, B*N)
      const int col = ct * 16 + lo;           // head dim in [0,64)
      const _Float16 h = (_Float16)c[ct][r];
      if (mat == 2) {
        const int bb = grow >> 12;  // N = 4096
        const int n = grow & 4095;
        vt[(size_t)(bb * 64 + col) * 4096 + n] = h;
      } else if (mat == 0) {
        qh[(size_t)grow * 64 + col] = h;
      } else {
        kh[(size_t)grow * 64 + col] = h;
      }
    }
  }
}

// ---------------------------------------------------------------------------
// Kernel 2: split-K flash attention.
// Block = 256 threads = 8 waves = 2 query tiles x 4 key-range splits.
// Each wave runs an independent online-softmax over 1024 keys (32-key WMMA
// chunks, ping-ponged K fragments), then the block merges partial (M, L, O)
// states through LDS with the standard flash combine. 4096 waves total.
// ---------------------------------------------------------------------------
struct KF {
  v16h f[4];
};

static __device__ __forceinline__ void load_kf(const _Float16* kbat, int kb,
                                               int lo, int hi, KF& k) {
  const _Float16* kp0 = kbat + (size_t)(kb + lo) * 64 + hi * 16;
  k.f[0] = *(const v16h*)(kp0);
  k.f[1] = *(const v16h*)(kp0 + 32);
  k.f[2] = *(const v16h*)(kp0 + 16 * 64);
  k.f[3] = *(const v16h*)(kp0 + 16 * 64 + 32);
}

static __device__ __forceinline__ void attn_step32(
    const v16h a0, const v16h a1, const KF& kf, KF& kfn,
    const _Float16* __restrict__ kbat, const _Float16* __restrict__ vbat,
    int kb, int kbn, int lo, int hi, _Float16* pw, float M[8], float L[8],
    v8f& O0, v8f& O1, v8f& O2, v8f& O3) {
  // V B-fragments for this chunk, issued first; consumed only after softmax
  // so their latency hides under 4 WMMAs + the whole softmax VALU chain.
  const _Float16* vp = vbat + kb + hi * 16;
  v16h bv0 = *(const v16h*)(vp + (size_t)(lo)*4096);
  v16h bv1 = *(const v16h*)(vp + (size_t)(16 + lo) * 4096);
  v16h bv2 = *(const v16h*)(vp + (size_t)(32 + lo) * 4096);
  v16h bv3 = *(const v16h*)(vp + (size_t)(48 + lo) * 4096);
  SCHED_FENCE();

  v8f s0 = {}, s1 = {};
  s0 = wmma_f16(a0, kf.f[0], s0);
  s0 = wmma_f16(a1, kf.f[1], s0);
  s1 = wmma_f16(a0, kf.f[2], s1);
  s1 = wmma_f16(a1, kf.f[3], s1);

  // pipeline: next chunk's K fragments into the other buffer
  load_kf(kbat, kbn, lo, hi, kfn);
  SCHED_FENCE();

  // Online softmax. Row of reg r = r + 8*hi; cols live across the 16-lane
  // half, so xor-1/2/4/8 butterflies reduce exactly one row.
  float alpha[8];
#pragma unroll
  for (int r = 0; r < 8; ++r) {
    float x0 = s0[r] * 0.125f;  // 1/sqrt(64)
    float x1 = s1[r] * 0.125f;
    float t = fmaxf(x0, x1);
    t = fmaxf(t, __shfl_xor(t, 1, 32));
    t = fmaxf(t, __shfl_xor(t, 2, 32));
    t = fmaxf(t, __shfl_xor(t, 4, 32));
    t = fmaxf(t, __shfl_xor(t, 8, 32));
    const float mn = fmaxf(M[r], t);
    const float al = __expf(M[r] - mn);
    M[r] = mn;
    const float p0 = __expf(x0 - mn);
    const float p1 = __expf(x1 - mn);
    s0[r] = p0;
    s1[r] = p1;
    float sm = p0 + p1;
    sm += __shfl_xor(sm, 1, 32);
    sm += __shfl_xor(sm, 2, 32);
    sm += __shfl_xor(sm, 4, 32);
    sm += __shfl_xor(sm, 8, 32);
    L[r] = L[r] * al + sm;
    alpha[r] = al;
  }
#pragma unroll
  for (int r = 0; r < 8; ++r) {
    O0[r] *= alpha[r];
    O1[r] *= alpha[r];
    O2[r] *= alpha[r];
    O3[r] *= alpha[r];
  }

  // Relayout P: D-layout regs -> LDS (row-major 16x32 f16) -> A-layout frag
#pragma unroll
  for (int r = 0; r < 8; ++r) {
    const int row = r + hi * 8;
    pw[row * 32 + lo] = (_Float16)s0[r];
    pw[row * 32 + 16 + lo] = (_Float16)s1[r];
  }
  asm volatile("s_wait_dscnt 0x0" ::: "memory");
  const _Float16* pr = pw + lo * 32 + hi * 16;
  v16h pa = cat8(*(const v8h*)(pr), *(const v8h*)(pr + 8));
  asm volatile("s_wait_dscnt 0x0" ::: "memory");

  O0 = wmma_f16(pa, bv0, O0);
  O1 = wmma_f16(pa, bv1, O1);
  O2 = wmma_f16(pa, bv2, O2);
  O3 = wmma_f16(pa, bv3, O3);
}

__global__ __launch_bounds__(256) void attn_kernel(
    const _Float16* __restrict__ qh, const _Float16* __restrict__ kh,
    const _Float16* __restrict__ vt, float* __restrict__ out) {
  __shared__ _Float16 plds[8][512];   // per-wave P staging: 8 KB
  __shared__ float olds[8][16][64];   // per-wave partial O: 32 KB
  __shared__ float mlds[8][16];       // per-wave row maxes
  __shared__ float llds[8][16];       // per-wave row sums

  const int wave = threadIdx.x >> 5;
  const int lane = threadIdx.x & 31;
  const int hi = lane >> 4, lo = lane & 15;
  const int g = wave >> 2;               // query tile within block (0/1)
  const int sp = wave & 3;               // key-range split (0..3)
  const int qt = blockIdx.x * 2 + g;     // 0..1023 query tiles
  const int b = qt >> 8;                 // 256 tiles per batch

  // Q A-fragments (persistent): a0 covers h=0..31, a1 covers h=32..63
  const _Float16* qp = qh + (size_t)(qt * 16 + lo) * 64;
  const v16h a0 =
      cat8(*(const v8h*)(qp + hi * 16), *(const v8h*)(qp + 8 + hi * 16));
  const v16h a1 =
      cat8(*(const v8h*)(qp + 32 + hi * 16), *(const v8h*)(qp + 40 + hi * 16));

  float M[8], L[8];
#pragma unroll
  for (int r = 0; r < 8; ++r) {
    M[r] = -1e30f;
    L[r] = 0.0f;
  }
  v8f O0 = {}, O1 = {}, O2 = {}, O3 = {};

  const _Float16* kbat = kh + (size_t)b * 4096 * 64;
  const _Float16* vbat = vt + (size_t)b * 64 * 4096;
  _Float16* pw = plds[wave];

  const int kb0 = sp * 1024;  // this wave's key range: [kb0, kb0+1024)
  const int kb1 = kb0 + 1024;

  KF kA, kB;
  load_kf(kbat, kb0, lo, hi, kA);

  for (int kb = kb0; kb < kb1; kb += 64) {
    attn_step32(a0, a1, kA, kB, kbat, vbat, kb, kb + 32, lo, hi, pw, M, L, O0,
                O1, O2, O3);
    const int kn2 = (kb + 64 < kb1) ? kb + 64 : kb0;  // last load unused
    attn_step32(a0, a1, kB, kA, kbat, vbat, kb + 32, kn2, lo, hi, pw, M, L, O0,
                O1, O2, O3);
  }

  // Publish partial state to LDS
#pragma unroll
  for (int r = 0; r < 8; ++r) {
    const int row = r + hi * 8;
    olds[wave][row][lo] = O0[r];
    olds[wave][row][16 + lo] = O1[r];
    olds[wave][row][32 + lo] = O2[r];
    olds[wave][row][48 + lo] = O3[r];
  }
  if (lo == 0) {
#pragma unroll
    for (int r = 0; r < 8; ++r) {
      mlds[wave][r + hi * 8] = M[r];
      llds[wave][r + hi * 8] = L[r];
    }
  }
  __syncthreads();

  // Flash combine of 4 splits per query tile; fully coalesced f32 stores.
  // 2048 outputs per block, 256 threads -> 8 outputs each.
#pragma unroll
  for (int j = 0; j < 8; ++j) {
    const int f = threadIdx.x + 256 * j;  // 0..2047
    const int gg = f >> 10;               // query tile in block
    const int row = (f >> 6) & 15;
    const int col = f & 63;
    const int w0 = gg * 4;
    const float m0 = mlds[w0][row], m1 = mlds[w0 + 1][row];
    const float m2 = mlds[w0 + 2][row], m3 = mlds[w0 + 3][row];
    const float mg = fmaxf(fmaxf(m0, m1), fmaxf(m2, m3));
    const float e0 = __expf(m0 - mg), e1 = __expf(m1 - mg);
    const float e2 = __expf(m2 - mg), e3 = __expf(m3 - mg);
    const float denom = llds[w0][row] * e0 + llds[w0 + 1][row] * e1 +
                        llds[w0 + 2][row] * e2 + llds[w0 + 3][row] * e3;
    const float num = olds[w0][row][col] * e0 + olds[w0 + 1][row][col] * e1 +
                      olds[w0 + 2][row][col] * e2 + olds[w0 + 3][row][col] * e3;
    out[(size_t)((blockIdx.x * 2 + gg) * 16 + row) * 64 + col] = num / denom;
  }
}

extern "C" void kernel_launch(void* const* d_in, const int* in_sizes, int n_in,
                              void* d_out, int out_size, void* d_ws,
                              size_t ws_size, hipStream_t stream) {
  (void)in_sizes; (void)n_in; (void)out_size; (void)ws_size;
  const float* x = (const float*)d_in[0];
  const float* Wk = (const float*)d_in[1];
  const float* Wq = (const float*)d_in[2];
  const float* Wv = (const float*)d_in[3];
  float* out = (float*)d_out;

  _Float16* qh = (_Float16*)d_ws;                // [B*N, 64] f16 : 2 MB
  _Float16* kh = qh + (size_t)4 * 4096 * 64;     // [B*N, 64] f16 : 2 MB
  _Float16* vt = kh + (size_t)4 * 4096 * 64;     // [B, 64, N] f16: 2 MB
  _Float16* xh = vt + (size_t)4 * 4096 * 64;     // [B*N, 1024] f16: 32 MB
  _Float16* wqh = xh + (size_t)4 * 4096 * 1024;  // [64,1024] f16: 128 KB
  _Float16* wkh = wqh + (size_t)64 * 1024;
  _Float16* wvh = wkh + (size_t)64 * 1024;

  // Stage 0: one-pass f32 -> f16 conversion (bandwidth bound)
  cvt_f16_kernel<<<dim3(2048), dim3(256), 0, stream>>>(x, xh,
                                                       4 * 4096 * 1024 / 8);
  cvt_f16_kernel<<<dim3(32), dim3(256), 0, stream>>>(Wq, wqh, 64 * 1024 / 8);
  cvt_f16_kernel<<<dim3(32), dim3(256), 0, stream>>>(Wk, wkh, 64 * 1024 / 8);
  cvt_f16_kernel<<<dim3(32), dim3(256), 0, stream>>>(Wv, wvh, 64 * 1024 / 8);

  // Stage 1: QKV projection (pure f16 WMMA)
  qkv_proj_kernel<<<dim3(128, 3), dim3(256), 0, stream>>>(xh, wqh, wkh, wvh, qh,
                                                          kh, vt);
  // Stage 2: split-K flash attention (4 waves per query tile)
  attn_kernel<<<dim3(512), dim3(256), 0, stream>>>(qh, kh, vt, out);
}
